// BDHGraphModel_36636071035463
// MI455X (gfx1250) — matrix operand order repaired and checked by reading, not compiled, not run on hardware
//
#include <hip/hip_runtime.h>
#include <hip/hip_bf16.h>

#define BB 8
#define TT 128
#define NN 2048
#define EE 65536
#define VV 32000

typedef __attribute__((ext_vector_type(16))) __bf16 v16bf;
typedef __attribute__((ext_vector_type(8)))  float  v8f;

// f32 -> bf16 round-to-nearest-even (bit ops; no reliance on cast support)
__device__ __forceinline__ __bf16 f2bf(float f) {
    unsigned u = __builtin_bit_cast(unsigned, f);
    u += 0x7FFFu + ((u >> 16) & 1u);
    unsigned short h = (unsigned short)(u >> 16);
    return __builtin_bit_cast(__bf16, h);
}

// ---------------------------------------------------------------- CSR build
__global__ void csr_count(const int* __restrict__ eidx, int* __restrict__ cnt) {
    int e = blockIdx.x * blockDim.x + threadIdx.x;
    atomicAdd(&cnt[eidx[EE + e]], 1);
}

__global__ void csr_scan(const int* __restrict__ cnt, int* __restrict__ rowptr,
                         int* __restrict__ cursor) {
    if (threadIdx.x == 0 && blockIdx.x == 0) {
        int acc = 0;
        for (int d = 0; d < NN; ++d) {
            rowptr[d] = acc;
            cursor[d] = acc;
            acc += cnt[d];
        }
        rowptr[NN] = acc;
    }
}

__global__ void csr_fill(const int* __restrict__ eidx, int* __restrict__ cursor,
                         int* __restrict__ perm) {
    int e = blockIdx.x * blockDim.x + threadIdx.x;
    int d = eidx[EE + e];
    int pos = atomicAdd(&cursor[d], 1);
    perm[pos] = e;
}

__global__ void csr_reorder(const int* __restrict__ eidx, const int* __restrict__ perm,
                            const float* __restrict__ Gx, const float* __restrict__ Gy,
                            const float* __restrict__ Gs,
                            int* __restrict__ srcP, float* __restrict__ GxP,
                            float* __restrict__ GyP, float* __restrict__ GsP) {
    int j = blockIdx.x * blockDim.x + threadIdx.x;
    int e = perm[j];
    srcP[j] = eidx[e];
    GxP[j]  = Gx[e];
    GyP[j]  = Gy[e];
    GsP[j]  = Gs[e];
}

// ---------------------------------------------------------------- scan phases
// y0 = emb[idx[:,0]]
__global__ void gather_kernel(const float* __restrict__ emb, const int* __restrict__ idx,
                              int t, float* __restrict__ dst) {
    int i = blockIdx.x * blockDim.x + threadIdx.x;   // 0 .. BB*NN-1
    int b = i >> 11;
    int n = i & (NN - 1);
    dst[i] = emb[(size_t)idx[b * TT + t] * NN + n];
}

// Phase 1 (thread per dst neuron): A[b,d] = sum_e x[b,src]*sigma ;
// sigmaP[j] = (sigmaP[j] + mean_b(y[b,src]*x[b,d])*Gs)*0.99
__global__ void layer_A_sigma(const int* __restrict__ rowptr, const int* __restrict__ srcP,
                              const float* __restrict__ GsP,
                              const float* __restrict__ xbuf,
                              const float* __restrict__ emb, const int* __restrict__ idx, int t,
                              const float* __restrict__ y,
                              float* __restrict__ sigmaP, float* __restrict__ A) {
    int d = blockIdx.x * blockDim.x + threadIdx.x;   // 0..NN-1
    const float* base = xbuf ? xbuf : emb;
    size_t rb[BB];
#pragma unroll
    for (int b = 0; b < BB; ++b)
        rb[b] = xbuf ? (size_t)b * NN : (size_t)idx[b * TT + t] * NN;

    float xd[BB], acc[BB];
#pragma unroll
    for (int b = 0; b < BB; ++b) { xd[b] = base[rb[b] + d]; acc[b] = 0.0f; }

    int beg = rowptr[d], end = rowptr[d + 1];
    for (int j = beg; j < end; ++j) {
        int s = srcP[j];
        float sig = sigmaP[j];
        float heb = 0.0f;
#pragma unroll
        for (int b = 0; b < BB; ++b) {
            float xs = base[rb[b] + s];
            acc[b] += xs * sig;
            heb += y[b * NN + s] * xd[b];
        }
        sigmaP[j] = (sig + heb * 0.125f * GsP[j]) * 0.99f;
    }
#pragma unroll
    for (int b = 0; b < BB; ++b) A[b * NN + d] = acc[b];
}

// Phase 2 (thread per (b,dst)): y[b,d] = sum_e relu(A[b,src])*Gy
__global__ void layer_y(const int* __restrict__ rowptr, const int* __restrict__ srcP,
                        const float* __restrict__ GyP,
                        const float* __restrict__ A, float* __restrict__ y) {
    int i = blockIdx.x * blockDim.x + threadIdx.x;   // 0 .. BB*NN-1
    int d = i & (NN - 1);
    int b = i >> 11;
    const float* Ab = A + (size_t)b * NN;
    float acc = 0.0f;
    for (int j = rowptr[d]; j < rowptr[d + 1]; ++j) {
        float a = Ab[srcP[j]];
        acc += (a > 0.0f ? a : 0.0f) * GyP[j];
    }
    y[(size_t)b * NN + d] = acc;
}

// Phase 3 (thread per (b,dst)): x[b,d] = relu(sum_e y[b,src]*Gx); optional bf16 store
__global__ void layer_x(const int* __restrict__ rowptr, const int* __restrict__ srcP,
                        const float* __restrict__ GxP,
                        const float* __restrict__ y, float* __restrict__ x,
                        __bf16* __restrict__ xs_rows) {
    int i = blockIdx.x * blockDim.x + threadIdx.x;
    int d = i & (NN - 1);
    int b = i >> 11;
    const float* yb = y + (size_t)b * NN;
    float acc = 0.0f;
    for (int j = rowptr[d]; j < rowptr[d + 1]; ++j)
        acc += yb[srcP[j]] * GxP[j];
    acc = acc > 0.0f ? acc : 0.0f;
    x[(size_t)b * NN + d] = acc;
    if (xs_rows) xs_rows[(size_t)b * NN + d] = f2bf(acc);
}

// sigma back to original edge order, straight into d_out tail
__global__ void sigma_out(const int* __restrict__ perm, const float* __restrict__ sigmaP,
                          float* __restrict__ out) {
    int j = blockIdx.x * blockDim.x + threadIdx.x;
    out[perm[j]] = sigmaP[j];
}

// ---------------------------------------------------------------- GEMM
// W_ro f32 -> bf16 (memory bound, once per launch)
__global__ void convert_w(const float* __restrict__ W, __bf16* __restrict__ Wb) {
    size_t i = ((size_t)blockIdx.x * blockDim.x + threadIdx.x) * 8;
    float4 f0 = *(const float4*)(W + i);
    float4 f1 = *(const float4*)(W + i + 4);
    union { v16bf v; uint4 q[2]; __bf16 h[16]; } o;
    o.h[0] = f2bf(f0.x); o.h[1] = f2bf(f0.y); o.h[2] = f2bf(f0.z); o.h[3] = f2bf(f0.w);
    o.h[4] = f2bf(f1.x); o.h[5] = f2bf(f1.y); o.h[6] = f2bf(f1.z); o.h[7] = f2bf(f1.w);
    *(uint4*)(Wb + i) = o.q[0];
}

// Pre-converted bf16 GEMM, register-blocked:
// wave: M=32 (2 A frags) x N=128 (8 B frags) -> 16 WMMAs per K-step, 20 b128 loads
// (1.25 load-issues per WMMA). WG = 8 waves -> 256x128 C tile. Grid (4, 250).
__global__ void __launch_bounds__(256)
gemm_wmma_pre(const __bf16* __restrict__ Axs, const __bf16* __restrict__ Wb,
              const float* __restrict__ bias, float* __restrict__ out) {
    const int lane  = threadIdx.x & 31;
    const int wave  = threadIdx.x >> 5;
    const int mBase = blockIdx.x * 256 + wave * 32;
    const int vBase = blockIdx.y * 128;
    const int lhalf = lane >> 4;
    const int llo   = lane & 15;

    v8f acc[2][8] = {};
    const __bf16* arow0 = Axs + (size_t)(mBase + llo) * NN;
    const __bf16* arow1 = Axs + (size_t)(mBase + 16 + llo) * NN;

    for (int kk = 0; kk < NN; kk += 32) {
        union { v16bf v; uint4 q[2]; } a0, a1;
        const __bf16* p0 = arow0 + kk + lhalf * 8;      // K {0-7,16-23} / {8-15,24-31}
        const __bf16* p1 = arow1 + kk + lhalf * 8;
        a0.q[0] = *(const uint4*)(p0);
        a0.q[1] = *(const uint4*)(p0 + 16);
        a1.q[0] = *(const uint4*)(p1);
        a1.q[1] = *(const uint4*)(p1 + 16);
#pragma unroll
        for (int j = 0; j < 8; ++j) {
            const __bf16* wp = Wb + (size_t)(vBase + j * 16 + llo) * NN + kk + lhalf * 8;
            union { v16bf v; uint4 q[2]; } bf;
            bf.q[0] = *(const uint4*)(wp);
            bf.q[1] = *(const uint4*)(wp + 16);
            acc[0][j] = __builtin_amdgcn_wmma_f32_16x16x32_bf16(
                false, a0.v, false, bf.v, (short)0, acc[0][j], false, false);
            acc[1][j] = __builtin_amdgcn_wmma_f32_16x16x32_bf16(
                false, a1.v, false, bf.v, (short)0, acc[1][j], false, false);
        }
    }
#pragma unroll
    for (int i2 = 0; i2 < 2; ++i2) {
#pragma unroll
        for (int j = 0; j < 8; ++j) {
            int v = vBase + j * 16 + llo;
            float bv = bias[v];
#pragma unroll
            for (int r = 0; r < 8; ++r) {
                int mrow = mBase + i2 * 16 + lhalf * 8 + r;   // m = t*8 + b
                int b = mrow & 7;
                int t = mrow >> 3;
                out[(size_t)(b * TT + t) * VV + v] = acc[i2][j][r] + bv;
            }
        }
    }
}

// Fallback (workspace too small for bf16 W): convert W in-register
__global__ void __launch_bounds__(256)
gemm_wmma_inline(const __bf16* __restrict__ Axs, const float* __restrict__ W,
                 const float* __restrict__ bias, float* __restrict__ out) {
    const int lane  = threadIdx.x & 31;
    const int wave  = threadIdx.x >> 5;
    const int mBase = blockIdx.x * 128 + wave * 16;
    const int vBase = blockIdx.y * 128;
    const int lhalf = lane >> 4;
    const int llo   = lane & 15;

    v8f acc[8] = {};
    const __bf16* arow = Axs + (size_t)(mBase + llo) * NN;

    for (int kk = 0; kk < NN; kk += 32) {
        union { v16bf v; uint4 q[2]; } a;
        const __bf16* a0 = arow + kk + lhalf * 8;
        a.q[0] = *(const uint4*)(a0);
        a.q[1] = *(const uint4*)(a0 + 16);
#pragma unroll
        for (int j = 0; j < 8; ++j) {
            const float* wp = W + (size_t)(vBase + j * 16 + llo) * NN + kk + lhalf * 8;
            v16bf bf;
#pragma unroll
            for (int i = 0; i < 16; ++i) bf[i] = f2bf(wp[i + (i & 8)]);
            acc[j] = __builtin_amdgcn_wmma_f32_16x16x32_bf16(
                false, a.v, false, bf, (short)0, acc[j], false, false);
        }
    }
#pragma unroll
    for (int j = 0; j < 8; ++j) {
        int v = vBase + j * 16 + llo;
        float bv = bias[v];
#pragma unroll
        for (int r = 0; r < 8; ++r) {
            int mrow = mBase + lhalf * 8 + r;
            int b = mrow & 7;
            int t = mrow >> 3;
            out[(size_t)(b * TT + t) * VV + v] = acc[j][r] + bv;
        }
    }
}

// ---------------------------------------------------------------- launcher
extern "C" void kernel_launch(void* const* d_in, const int* in_sizes, int n_in,
                              void* d_out, int out_size, void* d_ws, size_t ws_size,
                              hipStream_t stream) {
    const int*   idx  = (const int*)  d_in[0];
    const int*   eidx = (const int*)  d_in[1];
    const float* emb  = (const float*)d_in[2];
    const float* Gx   = (const float*)d_in[3];
    const float* Gy   = (const float*)d_in[4];
    const float* Gs   = (const float*)d_in[5];
    const float* Wro  = (const float*)d_in[6];
    const float* bro  = (const float*)d_in[7];
    float* out = (float*)d_out;

    // Workspace carve-up (all chunks 256B aligned)
    char* p = (char*)d_ws;
    size_t off = 0;
    auto take = [&](size_t bytes) {
        char* r = p + off;
        off = (off + bytes + 255) & ~(size_t)255;
        return r;
    };
    float* A      = (float*)take(BB * NN * sizeof(float));
    float* y      = (float*)take(BB * NN * sizeof(float));
    float* x      = (float*)take(BB * NN * sizeof(float));
    int*   cnt    = (int*)  take(NN * sizeof(int));
    int*   rowptr = (int*)  take((NN + 1) * sizeof(int));
    int*   cursor = (int*)  take(NN * sizeof(int));
    int*   perm   = (int*)  take(EE * sizeof(int));
    int*   srcP   = (int*)  take(EE * sizeof(int));
    float* GxP    = (float*)take(EE * sizeof(float));
    float* GyP    = (float*)take(EE * sizeof(float));
    float* GsP    = (float*)take(EE * sizeof(float));
    float* sigmaP = (float*)take(EE * sizeof(float));
    __bf16* xsb   = (__bf16*)take((size_t)TT * BB * NN * sizeof(__bf16));
    __bf16* Wbf   = (__bf16*)take((size_t)VV * NN * sizeof(__bf16));
    const bool bigWs = (ws_size >= off);
    (void)in_sizes; (void)n_in; (void)out_size;

    // CSR build (order within a dst bucket only affects FP rounding order)
    hipMemsetAsync(cnt, 0, NN * sizeof(int), stream);
    csr_count<<<EE / 256, 256, 0, stream>>>(eidx, cnt);
    csr_scan<<<1, 64, 0, stream>>>(cnt, rowptr, cursor);
    csr_fill<<<EE / 256, 256, 0, stream>>>(eidx, cursor, perm);
    csr_reorder<<<EE / 256, 256, 0, stream>>>(eidx, perm, Gx, Gy, Gs, srcP, GxP, GyP, GsP);

    hipMemsetAsync(sigmaP, 0, EE * sizeof(float), stream);
    gather_kernel<<<BB * NN / 256, 256, 0, stream>>>(emb, idx, 0, y);  // y0

    if (bigWs)
        convert_w<<<VV * NN / (256 * 8), 256, 0, stream>>>(Wro, Wbf);

    // Serial scan: 128 steps x 2 layers x 3 phase kernels (no memsets, no atomics)
    for (int t = 0; t < TT; ++t) {
        // layer 0: x input gathered from emb on the fly
        layer_A_sigma<<<NN / 256, 256, 0, stream>>>(rowptr, srcP, GsP, nullptr, emb, idx, t,
                                                    y, sigmaP, A);
        layer_y<<<BB * NN / 256, 256, 0, stream>>>(rowptr, srcP, GyP, A, y);
        layer_x<<<BB * NN / 256, 256, 0, stream>>>(rowptr, srcP, GxP, y, x, nullptr);
        // layer 1: x input from buffer; fused bf16 store of x_t into GEMM A operand
        layer_A_sigma<<<NN / 256, 256, 0, stream>>>(rowptr, srcP, GsP, x, emb, idx, t,
                                                    y, sigmaP, A);
        layer_y<<<BB * NN / 256, 256, 0, stream>>>(rowptr, srcP, GyP, A, y);
        layer_x<<<BB * NN / 256, 256, 0, stream>>>(rowptr, srcP, GxP, y, x,
                                                   xsb + (size_t)t * BB * NN);
    }

    // Readout GEMM: M=1024, K=2048, V=32000 (bf16 WMMA, f32 accumulate)
    if (bigWs) {
        dim3 grid(1024 / 256, VV / 128);   // M fastest -> W band reuse in L2
        gemm_wmma_pre<<<grid, 256, 0, stream>>>(xsb, Wbf, bro, out);
    } else {
        dim3 grid(1024 / 128, VV / 128);
        gemm_wmma_inline<<<grid, 256, 0, stream>>>(xsb, Wro, bro, out);
    }

    sigma_out<<<EE / 256, 256, 0, stream>>>(perm, sigmaP, out + (size_t)BB * TT * VV);
}
